// RelationshipAttention_55911884259606
// MI455X (gfx1250) — matrix operand order repaired and checked by reading, not compiled, not run on hardware
//
#include <hip/hip_runtime.h>
#include <hip/hip_bf16.h>
#include <math.h>

typedef __attribute__((ext_vector_type(16))) _Float16 v16h;
typedef __attribute__((ext_vector_type(8)))  float    v8f;

#define LN_EPS_F 1e-5f
#define CDIM 256
#define HID  512
#define NTOK 16384          // B*N
#define NB   8
#define NPER 2048           // N
#define TOPI 512
#define TOPK 16384
#define NPAIR (NB*TOPK)     // 131072
#define NIROW (NB*TOPI)     // 4096
#define SFLAT (TOPI*TOPI)   // 262144

// ---------------------------------------------------------------- helpers ---
__device__ __forceinline__ float gelu_exact(float x) {
  return 0.5f * x * (1.0f + erff(x * 0.70710678118654752f));
}

__device__ __forceinline__ float warp_sum(float v) {
#pragma unroll
  for (int m = 16; m >= 1; m >>= 1) v += __shfl_xor(v, m, 32);
  return v;
}

// A fragment (16x32 f16, MxK): lane<16 -> row=lane, K = k0+[0..15]
//                              lane>=16 -> row=lane-16, K = k0+[16..31]
__device__ __forceinline__ v16h load_frag_a(const _Float16* __restrict__ src,
                                            int ld, int lane, int k0) {
  const int row  = lane & 15;
  const int koff = (lane >> 4) * 16;
  const _Float16* p = src + row * ld + k0 + koff;
  v16h f;
#pragma unroll
  for (int i = 0; i < 8; ++i) f[i] = p[i];
#pragma unroll
  for (int i = 0; i < 8; ++i) f[8 + i] = p[8 + i];
  return f;
}

// B fragment (32x16 f16, KxN) built from row-major W[N][K]: B[k][n] = W[n][k]
// lane<16 -> n=lane, K = k0+[0..15]; lane>=16 -> n=lane-16, K = k0+[16..31]
__device__ __forceinline__ v16h load_frag_b(const _Float16* __restrict__ W,
                                            int ld, int lane, int k0) {
  const int n    = lane & 15;
  const int koff = (lane >> 4) * 16;
  const _Float16* p = W + n * ld + k0 + koff;
  v16h f;
#pragma unroll
  for (int i = 0; i < 16; ++i) f[i] = p[i];
  return f;
}

__device__ __forceinline__ v8f wmma16(v16h a, v16h b, v8f c) {
  return __builtin_amdgcn_wmma_f32_16x16x32_f16(false, a, false, b,
                                                (short)0, c, false, false);
}

__device__ __forceinline__ v8f zero8() {
  v8f z = {0.f, 0.f, 0.f, 0.f, 0.f, 0.f, 0.f, 0.f};
  return z;
}

__device__ __forceinline__ unsigned mono_key(float f) {
  unsigned u = __float_as_uint(f);
  return (u & 0x80000000u) ? ~u : (u | 0x80000000u);
}

__device__ __forceinline__ float softplus_tau(const float* tr) {
  return log1pf(expf(tr[0])) + 1e-8f;
}

// bitonic sort: (val desc, idx asc) total order
__device__ inline void bitonic_desc(float* sv, int* si, int n, int tid, int nth) {
  for (int k = 2; k <= n; k <<= 1) {
    for (int j = k >> 1; j > 0; j >>= 1) {
      for (int i = tid; i < n; i += nth) {
        int ixj = i ^ j;
        if (ixj > i) {
          bool  up = ((i & k) == 0);
          float a = sv[i], b = sv[ixj];
          int   ia = si[i], ib = si[ixj];
          bool agtb = (a > b) || (a == b && ia < ib);
          bool swp  = up ? (!agtb) : agtb;
          if (swp) { sv[i] = b; sv[ixj] = a; si[i] = ib; si[ixj] = ia; }
        }
      }
      __syncthreads();
    }
  }
}

// ---------------------------------------------------------------- kernels ---
__global__ __launch_bounds__(256)
void f32_to_f16_kernel(const float* __restrict__ src, _Float16* __restrict__ dst, int n) {
  int i = blockIdx.x * 256 + threadIdx.x;
  if (i < n) dst[i] = (_Float16)src[i];
}

// Residual MLP: out = LN(gelu(x@w1.T+b1)@wo.T+bo + x, g, beta). 64 rows / WG, 4 waves.
__global__ __launch_bounds__(128)
void resmlp_kernel(const float* __restrict__ x, const _Float16* __restrict__ w1,
                   const float* __restrict__ b1, const _Float16* __restrict__ wo,
                   const float* __restrict__ bo, const float* __restrict__ g,
                   const float* __restrict__ beta, float* __restrict__ out) {
  const int LDA = 264;
  extern __shared__ char smem[];
  _Float16* xs = (_Float16*)smem;              // [64][264]
  _Float16* hs = xs + 64 * LDA;                // [64][264]
  float*    ob = (float*)(hs + 64 * LDA);      // [64][256]

  const int tid = threadIdx.x, wave = tid >> 5, lane = tid & 31;
  const long row0 = (long)blockIdx.x * 64;
  const int m0 = wave * 16;

  for (int i = tid; i < 64 * CDIM; i += 128) {
    int r = i >> 8, c = i & 255;
    xs[r * LDA + c] = (_Float16)x[(row0 + r) * CDIM + c];
  }
  __syncthreads();

  // GEMM1 + gelu -> hs
  for (int nc = 0; nc < 2; ++nc) {
    v8f acc[8];
#pragma unroll
    for (int t = 0; t < 8; ++t) acc[t] = zero8();
    for (int ks = 0; ks < 8; ++ks) {
      v16h a = load_frag_a(xs + m0 * LDA, LDA, lane, ks * 32);
#pragma unroll
      for (int t = 0; t < 8; ++t) {
        v16h b = load_frag_b(w1 + (nc * 128 + t * 16) * CDIM, CDIM, lane, ks * 32);
        acc[t] = wmma16(a, b, acc[t]);
      }
    }
    const int rbase = m0 + (lane >> 4) * 8;
#pragma unroll
    for (int t = 0; t < 8; ++t) {
      int col = nc * 128 + t * 16 + (lane & 15);
      float bb = b1[col];
#pragma unroll
      for (int v = 0; v < 8; ++v)
        hs[(rbase + v) * LDA + col] = (_Float16)gelu_exact(acc[t][v] + bb);
    }
  }
  __syncthreads();

  // GEMM2 + bias + residual -> ob (f32)
  for (int nc = 0; nc < 2; ++nc) {
    v8f acc[8];
#pragma unroll
    for (int t = 0; t < 8; ++t) acc[t] = zero8();
    for (int ks = 0; ks < 8; ++ks) {
      v16h a = load_frag_a(hs + m0 * LDA, LDA, lane, ks * 32);
#pragma unroll
      for (int t = 0; t < 8; ++t) {
        v16h b = load_frag_b(wo + (nc * 128 + t * 16) * CDIM, CDIM, lane, ks * 32);
        acc[t] = wmma16(a, b, acc[t]);
      }
    }
    const int rbase = m0 + (lane >> 4) * 8;
#pragma unroll
    for (int t = 0; t < 8; ++t) {
      int col = nc * 128 + t * 16 + (lane & 15);
      float bb = bo[col];
#pragma unroll
      for (int v = 0; v < 8; ++v)
        ob[(rbase + v) * CDIM + col] =
            acc[t][v] + bb + x[(row0 + rbase + v) * CDIM + col];
    }
  }
  __syncthreads();

  // LayerNorm per row (wave handles its own 16 rows)
  for (int r = 0; r < 16; ++r) {
    const float* prow = ob + (m0 + r) * CDIM;
    float s = 0.f;
#pragma unroll
    for (int i = 0; i < 8; ++i) s += prow[lane + i * 32];
    float mean = warp_sum(s) * (1.0f / CDIM);
    float vs = 0.f;
#pragma unroll
    for (int i = 0; i < 8; ++i) { float d = prow[lane + i * 32] - mean; vs += d * d; }
    float inv = rsqrtf(warp_sum(vs) * (1.0f / CDIM) + LN_EPS_F);
    long orow = (row0 + m0 + r) * CDIM;
#pragma unroll
    for (int i = 0; i < 8; ++i) {
      int c = lane + i * 32;
      out[orow + c] = (prow[c] - mean) * inv * g[c] + beta[c];
    }
  }
}

// Per-token cosine score + inverse norms. One wave per token.
__global__ __launch_bounds__(256)
void score_kernel(const float* __restrict__ sub, const float* __restrict__ obj,
                  const float* __restrict__ tau_raw, float* __restrict__ diag_scores,
                  float* __restrict__ inv_sub, float* __restrict__ inv_obj) {
  int tok = blockIdx.x * 8 + (threadIdx.x >> 5);
  int lane = threadIdx.x & 31;
  const float* ps = sub + (long)tok * CDIM;
  const float* po = obj + (long)tok * CDIM;
  float dp = 0.f, ss = 0.f, oo = 0.f;
#pragma unroll
  for (int i = 0; i < 8; ++i) {
    float a = ps[lane + i * 32], b = po[lane + i * 32];
    dp += a * b; ss += a * a; oo += b * b;
  }
  dp = warp_sum(dp); ss = warp_sum(ss); oo = warp_sum(oo);
  if (lane == 0) {
    float ns = fmaxf(sqrtf(ss), 1e-12f), no = fmaxf(sqrtf(oo), 1e-12f);
    float tau = softplus_tau(tau_raw);
    diag_scores[tok] = dp / (ns * no) * tau;
    inv_sub[tok] = 1.0f / ns;
    inv_obj[tok] = 1.0f / no;
  }
}

// Exact top-512 of 2048 per batch via full bitonic sort in LDS.
__global__ __launch_bounds__(1024)
void topI_kernel(const float* __restrict__ scores, int* __restrict__ idxI,
                 float* __restrict__ idxI_f) {
  __shared__ float sv[NPER];
  __shared__ int   si[NPER];
  int b = blockIdx.x, tid = threadIdx.x;
  for (int i = tid; i < NPER; i += 1024) { sv[i] = scores[b * NPER + i]; si[i] = i; }
  __syncthreads();
  bitonic_desc(sv, si, NPER, tid, 1024);
  for (int i = tid; i < TOPI; i += 1024) {
    idxI[b * TOPI + i]   = si[i];
    idxI_f[b * TOPI + i] = (float)si[i];
  }
}

// Gather selected rows; emit f32 originals, f16 normalized, and diag_rel = LN(s+o).
__global__ __launch_bounds__(256)
void gatherI_kernel(const float* __restrict__ sub, const float* __restrict__ obj,
                    const float* __restrict__ invs, const float* __restrict__ invo,
                    const int* __restrict__ idxI, float* __restrict__ subIo,
                    float* __restrict__ objIo, _Float16* __restrict__ subIn,
                    _Float16* __restrict__ objIn, float* __restrict__ diag_rel) {
  int row = blockIdx.x * 8 + (threadIdx.x >> 5);   // [0, 4096)
  int lane = threadIdx.x & 31;
  int b = row >> 9;
  int t = idxI[row];
  long src = ((long)b * NPER + t) * CDIM;
  long dst = (long)row * CDIM;
  float is = invs[b * NPER + t], io = invo[b * NPER + t];
  float sv[8], ov[8];
  float sum = 0.f;
#pragma unroll
  for (int i = 0; i < 8; ++i) {
    int c = lane + i * 32;
    float s_ = sub[src + c], o_ = obj[src + c];
    subIo[dst + c] = s_;  objIo[dst + c] = o_;
    subIn[dst + c] = (_Float16)(s_ * is);
    objIn[dst + c] = (_Float16)(o_ * io);
    sv[i] = s_; ov[i] = o_;
    sum += s_ + o_;
  }
  float mean = warp_sum(sum) * (1.0f / CDIM);
  float var = 0.f;
#pragma unroll
  for (int i = 0; i < 8; ++i) { float d = sv[i] + ov[i] - mean; var += d * d; }
  float inv = rsqrtf(warp_sum(var) * (1.0f / CDIM) + LN_EPS_F);
#pragma unroll
  for (int i = 0; i < 8; ++i) {
    int c = lane + i * 32;
    diag_rel[dst + c] = (sv[i] + ov[i] - mean) * inv;
  }
}

// S[b,i,j] = tau * <sub_n[i], obj_n[j]>, diagonal = -inf.  64 rows per WG.
__global__ __launch_bounds__(128)
void sgemm_kernel(const _Float16* __restrict__ subIn, const _Float16* __restrict__ objIn,
                  const float* __restrict__ tau_raw, float* __restrict__ S) {
  int b = blockIdx.x >> 3, tile = blockIdx.x & 7;
  int wave = threadIdx.x >> 5, lane = threadIdx.x & 31;
  float tau = softplus_tau(tau_raw);
  const _Float16* A  = subIn + ((long)b * TOPI + tile * 64) * CDIM;
  const _Float16* Bm = objIn + (long)b * TOPI * CDIM;
  float* Sout = S + ((long)b * TOPI + tile * 64) * TOPI;
  const int m0 = wave * 16;
  for (int nc = 0; nc < 4; ++nc) {
    v8f acc[8];
#pragma unroll
    for (int t = 0; t < 8; ++t) acc[t] = zero8();
    for (int ks = 0; ks < 8; ++ks) {
      v16h a = load_frag_a(A + m0 * CDIM, CDIM, lane, ks * 32);
#pragma unroll
      for (int t = 0; t < 8; ++t) {
        v16h bf = load_frag_b(Bm + (nc * 128 + t * 16) * CDIM, CDIM, lane, ks * 32);
        acc[t] = wmma16(a, bf, acc[t]);
      }
    }
    const int rbase = m0 + (lane >> 4) * 8;
#pragma unroll
    for (int t = 0; t < 8; ++t) {
      int col = nc * 128 + t * 16 + (lane & 15);
#pragma unroll
      for (int v = 0; v < 8; ++v) {
        int ig = tile * 64 + rbase + v;
        float val = acc[t][v] * tau;
        if (ig == col) val = -INFINITY;
        Sout[(rbase + v) * TOPI + col] = val;
      }
    }
  }
}

// 4-pass radix threshold selection (descending) for top-16384 per batch.
__global__ __launch_bounds__(1024)
void topk_thresh_kernel(const float* __restrict__ S, unsigned* __restrict__ thresh,
                        int* __restrict__ eq_need) {
  __shared__ unsigned hist[256];
  __shared__ unsigned s_prefix;
  __shared__ int s_want;
  int b = blockIdx.x, tid = threadIdx.x;
  const float* Sb = S + (long)b * SFLAT;
  if (tid == 0) { s_prefix = 0u; s_want = TOPK; }
  __syncthreads();
  for (int shift = 24; shift >= 0; shift -= 8) {
    for (int i = tid; i < 256; i += 1024) hist[i] = 0u;
    __syncthreads();
    unsigned prefix = s_prefix;
    unsigned pmask  = (shift == 24) ? 0u : (0xFFFFFFFFu << (shift + 8));
    for (int i = tid; i < SFLAT; i += 1024) {
      unsigned k = mono_key(Sb[i]);
      if ((k & pmask) == (prefix & pmask)) atomicAdd(&hist[(k >> shift) & 255], 1u);
    }
    __syncthreads();
    if (tid == 0) {
      int want = s_want;
      unsigned cum = 0u;
      int dig = 0;
      for (int v = 255; v >= 0; --v) {
        if (cum + hist[v] >= (unsigned)want) { dig = v; s_want = want - (int)cum; break; }
        cum += hist[v];
      }
      s_prefix = prefix | ((unsigned)dig << shift);
    }
    __syncthreads();
  }
  if (tid == 0) { thresh[b] = s_prefix; eq_need[b] = s_want; }
}

__global__ __launch_bounds__(32)
void zero_cnt_kernel(int* __restrict__ cnt) {
  if (threadIdx.x < 16) cnt[threadIdx.x] = 0;
}

__global__ __launch_bounds__(256)
void topk_compact_kernel(const float* __restrict__ S, const unsigned* __restrict__ thresh,
                         const int* __restrict__ eq_need, int* __restrict__ sel,
                         int* __restrict__ cnt_gt, int* __restrict__ cnt_eq) {
  long gid = (long)blockIdx.x * 256 + threadIdx.x;   // 8*262144
  int b = (int)(gid >> 18);
  int i = (int)(gid & (SFLAT - 1));
  unsigned k = mono_key(S[gid]);
  unsigned T = thresh[b];
  if (k > T) {
    int p = atomicAdd(&cnt_gt[b], 1);
    sel[b * TOPK + p] = i;
  } else if (k == T) {
    int need = eq_need[b];
    int p = atomicAdd(&cnt_eq[b], 1);
    if (p < need) sel[b * TOPK + (TOPK - need) + p] = i;
  }
}

// Sort the 16384 selected pairs per batch in 128KB LDS; emit scores & pairs.
__global__ __launch_bounds__(1024)
void topk_sort_kernel(const float* __restrict__ S, const int* __restrict__ sel,
                      const int* __restrict__ idxI, float* __restrict__ rel_scores,
                      float* __restrict__ rel_pairs, int* __restrict__ pair_out) {
  extern __shared__ char smem[];
  float* sv = (float*)smem;          // 16384 f32
  int*   si = (int*)(sv + TOPK);     // 16384 i32
  int b = blockIdx.x, tid = threadIdx.x;
  for (int i = tid; i < TOPK; i += 1024) {
    int p = sel[b * TOPK + i];
    si[i] = p;
    sv[i] = S[(long)b * SFLAT + p];
  }
  __syncthreads();
  bitonic_desc(sv, si, TOPK, tid, 1024);
  for (int i = tid; i < TOPK; i += 1024) {
    int p = si[i];
    int pi = p >> 9, pj = p & (TOPI - 1);
    pair_out[b * TOPK + i] = p;
    rel_scores[b * TOPK + i] = sv[i];
    rel_pairs[((long)b * TOPK + i) * 2 + 0] = (float)idxI[b * TOPI + pi];
    rel_pairs[((long)b * TOPK + i) * 2 + 1] = (float)idxI[b * TOPI + pj];
  }
}

// h = LN(LN(sub_sel+obj_sel), g, beta) -> f16.  One wave per pair row.
__global__ __launch_bounds__(256)
void pair_prep_kernel(const float* __restrict__ subIo, const float* __restrict__ objIo,
                      const int* __restrict__ pair_idx, const float* __restrict__ g,
                      const float* __restrict__ beta, _Float16* __restrict__ hsel) {
  int row = blockIdx.x * 8 + (threadIdx.x >> 5);   // [0, 131072)
  int lane = threadIdx.x & 31;
  int b = row >> 14;
  int p = pair_idx[row];
  int pi = p >> 9, pj = p & (TOPI - 1);
  const float* ps = subIo + ((long)b * TOPI + pi) * CDIM;
  const float* po = objIo + ((long)b * TOPI + pj) * CDIM;
  float vals[8];
  float sum = 0.f;
#pragma unroll
  for (int i = 0; i < 8; ++i) {
    int c = lane + i * 32;
    vals[i] = ps[c] + po[c];
    sum += vals[i];
  }
  float mean = warp_sum(sum) * (1.0f / CDIM);
  float var = 0.f;
#pragma unroll
  for (int i = 0; i < 8; ++i) { float d = vals[i] - mean; var += d * d; }
  float inv = rsqrtf(warp_sum(var) * (1.0f / CDIM) + LN_EPS_F);
#pragma unroll
  for (int i = 0; i < 8; ++i) vals[i] = (vals[i] - mean) * inv;
  // second LN (affine)
  sum = 0.f;
#pragma unroll
  for (int i = 0; i < 8; ++i) sum += vals[i];
  mean = warp_sum(sum) * (1.0f / CDIM);
  var = 0.f;
#pragma unroll
  for (int i = 0; i < 8; ++i) { float d = vals[i] - mean; var += d * d; }
  inv = rsqrtf(warp_sum(var) * (1.0f / CDIM) + LN_EPS_F);
#pragma unroll
  for (int i = 0; i < 8; ++i) {
    int c = lane + i * 32;
    hsel[(long)row * CDIM + c] = (_Float16)((vals[i] - mean) * inv * g[c] + beta[c]);
  }
}

// rel = gelu(h @ w1.T + b1) @ w2.T + b2.  64 rows / WG, hidden in LDS.
__global__ __launch_bounds__(128)
void postmlp_kernel(const _Float16* __restrict__ hsel, const _Float16* __restrict__ w1,
                    const float* __restrict__ b1, const _Float16* __restrict__ w2,
                    const float* __restrict__ b2, float* __restrict__ rel) {
  const int LDH = 520;
  extern __shared__ char smem[];
  _Float16* hs = (_Float16*)smem;   // [64][520]
  int wave = threadIdx.x >> 5, lane = threadIdx.x & 31;
  long row0 = (long)blockIdx.x * 64;
  int m0 = wave * 16;
  const _Float16* A = hsel + (row0 + m0) * CDIM;

  // GEMM1: 256 -> 512, gelu, into LDS
  for (int nc = 0; nc < 4; ++nc) {
    v8f acc[8];
#pragma unroll
    for (int t = 0; t < 8; ++t) acc[t] = zero8();
    for (int ks = 0; ks < 8; ++ks) {
      v16h a = load_frag_a(A, CDIM, lane, ks * 32);
#pragma unroll
      for (int t = 0; t < 8; ++t) {
        v16h bf = load_frag_b(w1 + (nc * 128 + t * 16) * CDIM, CDIM, lane, ks * 32);
        acc[t] = wmma16(a, bf, acc[t]);
      }
    }
    const int rbase = m0 + (lane >> 4) * 8;
#pragma unroll
    for (int t = 0; t < 8; ++t) {
      int col = nc * 128 + t * 16 + (lane & 15);
      float bb = b1[col];
#pragma unroll
      for (int v = 0; v < 8; ++v)
        hs[(rbase + v) * LDH + col] = (_Float16)gelu_exact(acc[t][v] + bb);
    }
  }
  __syncthreads();

  // GEMM2: 512 -> 256
  for (int nc = 0; nc < 2; ++nc) {
    v8f acc[8];
#pragma unroll
    for (int t = 0; t < 8; ++t) acc[t] = zero8();
    for (int ks = 0; ks < 16; ++ks) {
      v16h a = load_frag_a(hs + m0 * LDH, LDH, lane, ks * 32);
#pragma unroll
      for (int t = 0; t < 8; ++t) {
        v16h bf = load_frag_b(w2 + (nc * 128 + t * 16) * HID, HID, lane, ks * 32);
        acc[t] = wmma16(a, bf, acc[t]);
      }
    }
    const int rbase = m0 + (lane >> 4) * 8;
#pragma unroll
    for (int t = 0; t < 8; ++t) {
      int col = nc * 128 + t * 16 + (lane & 15);
      float bb = b2[col];
#pragma unroll
      for (int v = 0; v < 8; ++v)
        rel[(row0 + rbase + v) * CDIM + col] = acc[t][v] + bb;
    }
  }
}

// ----------------------------------------------------------------- launch ---
extern "C" void kernel_launch(void* const* d_in, const int* in_sizes, int n_in,
                              void* d_out, int out_size, void* d_ws, size_t ws_size,
                              hipStream_t stream) {
  (void)in_sizes; (void)n_in; (void)out_size; (void)ws_size;

  const float* tokens   = (const float*)d_in[0];
  const float* tau_raw  = (const float*)d_in[1];
  const float* sub_w1   = (const float*)d_in[2];
  const float* sub_b1   = (const float*)d_in[3];
  const float* sub_wo   = (const float*)d_in[4];
  const float* sub_bo   = (const float*)d_in[5];
  const float* sub_g    = (const float*)d_in[6];
  const float* sub_beta = (const float*)d_in[7];
  const float* obj_w1   = (const float*)d_in[8];
  const float* obj_b1   = (const float*)d_in[9];
  const float* obj_wo   = (const float*)d_in[10];
  const float* obj_bo   = (const float*)d_in[11];
  const float* obj_g    = (const float*)d_in[12];
  const float* obj_beta = (const float*)d_in[13];
  const float* post_g   = (const float*)d_in[14];
  const float* post_beta= (const float*)d_in[15];
  const float* post_w1  = (const float*)d_in[16];
  const float* post_b1  = (const float*)d_in[17];
  const float* post_w2  = (const float*)d_in[18];
  const float* post_b2  = (const float*)d_in[19];

  // workspace bump allocator
  char* ws = (char*)d_ws;
  size_t off = 0;
  auto alloc = [&](size_t bytes) -> void* {
    void* p = ws + off;
    off = (off + bytes + 255) & ~(size_t)255;
    return p;
  };

  _Float16* w1s = (_Float16*)alloc(CDIM * CDIM * 2);
  _Float16* wos = (_Float16*)alloc(CDIM * CDIM * 2);
  _Float16* w1o = (_Float16*)alloc(CDIM * CDIM * 2);
  _Float16* woo = (_Float16*)alloc(CDIM * CDIM * 2);
  _Float16* pw1 = (_Float16*)alloc(HID * CDIM * 2);
  _Float16* pw2 = (_Float16*)alloc(CDIM * HID * 2);
  float* subB   = (float*)alloc((size_t)NTOK * CDIM * 4);
  float* objB   = (float*)alloc((size_t)NTOK * CDIM * 4);
  float* invS   = (float*)alloc(NTOK * 4);
  float* invO   = (float*)alloc(NTOK * 4);
  int*   idxI   = (int*)alloc(NIROW * 4);
  float* subIo  = (float*)alloc((size_t)NIROW * CDIM * 4);
  float* objIo  = (float*)alloc((size_t)NIROW * CDIM * 4);
  _Float16* subIn = (_Float16*)alloc((size_t)NIROW * CDIM * 2);
  _Float16* objIn = (_Float16*)alloc((size_t)NIROW * CDIM * 2);
  float* Smat   = (float*)alloc((size_t)NB * SFLAT * 4);
  unsigned* thresh = (unsigned*)alloc(NB * 4);
  int* eq_need  = (int*)alloc(NB * 4);
  int* cnt      = (int*)alloc(16 * 4);        // [0..7]=gt, [8..15]=eq
  int* sel      = (int*)alloc((size_t)NPAIR * 4);
  int* pair_idx = (int*)alloc((size_t)NPAIR * 4);
  _Float16* hsel = (_Float16*)alloc((size_t)NPAIR * CDIM * 2);

  // d_out layout (floats, reference return order)
  float* out = (float*)d_out;
  float* o_rel        = out;                        // 8*16384*256
  float* o_pairs      = o_rel + (size_t)NPAIR * CDIM;
  float* o_scores     = o_pairs + (size_t)NPAIR * 2;
  float* o_diagrel    = o_scores + NPAIR;
  float* o_idxI       = o_diagrel + (size_t)NIROW * CDIM;
  float* o_diagscores = o_idxI + NIROW;

  // 1) weights -> f16
  f32_to_f16_kernel<<<256, 256, 0, stream>>>(sub_w1, w1s, CDIM * CDIM);
  f32_to_f16_kernel<<<256, 256, 0, stream>>>(sub_wo, wos, CDIM * CDIM);
  f32_to_f16_kernel<<<256, 256, 0, stream>>>(obj_w1, w1o, CDIM * CDIM);
  f32_to_f16_kernel<<<256, 256, 0, stream>>>(obj_wo, woo, CDIM * CDIM);
  f32_to_f16_kernel<<<512, 256, 0, stream>>>(post_w1, pw1, HID * CDIM);
  f32_to_f16_kernel<<<512, 256, 0, stream>>>(post_w2, pw2, CDIM * HID);

  // 2) residual MLPs (WMMA)
  const size_t RES_SHMEM = (size_t)64 * 264 * 2 * 2 + (size_t)64 * 256 * 4; // 133120
  resmlp_kernel<<<NTOK / 64, 128, RES_SHMEM, stream>>>(
      tokens, w1s, sub_b1, wos, sub_bo, sub_g, sub_beta, subB);
  resmlp_kernel<<<NTOK / 64, 128, RES_SHMEM, stream>>>(
      tokens, w1o, obj_b1, woo, obj_bo, obj_g, obj_beta, objB);

  // 3) cosine scores + norms
  score_kernel<<<NTOK / 8, 256, 0, stream>>>(subB, objB, tau_raw,
                                             o_diagscores, invS, invO);

  // 4) top-I per batch
  topI_kernel<<<NB, 1024, 0, stream>>>(o_diagscores, idxI, o_idxI);

  // 5) gather selected rows + diag_rel
  gatherI_kernel<<<NIROW / 8, 256, 0, stream>>>(subB, objB, invS, invO, idxI,
                                                subIo, objIo, subIn, objIn, o_diagrel);

  // 6) S = tau * sub_n @ obj_n^T with diag mask (WMMA)
  sgemm_kernel<<<NB * 8, 128, 0, stream>>>(subIn, objIn, tau_raw, Smat);

  // 7) top-K pair selection
  zero_cnt_kernel<<<1, 32, 0, stream>>>(cnt);
  topk_thresh_kernel<<<NB, 1024, 0, stream>>>(Smat, thresh, eq_need);
  topk_compact_kernel<<<(NB * SFLAT) / 256, 256, 0, stream>>>(
      Smat, thresh, eq_need, sel, cnt, cnt + 8);
  topk_sort_kernel<<<NB, 1024, (size_t)TOPK * 8, stream>>>(
      Smat, sel, idxI, o_scores, o_pairs, pair_idx);

  // 8) pair features: double LN -> f16
  pair_prep_kernel<<<NPAIR / 8, 256, 0, stream>>>(subIo, objIo, pair_idx,
                                                  post_g, post_beta, hsel);

  // 9) post MLP (WMMA) -> rel
  const size_t POST_SHMEM = (size_t)64 * 520 * 2;   // 66560
  postmlp_kernel<<<NPAIR / 64, 128, POST_SHMEM, stream>>>(
      hsel, pw1, post_b1, pw2, post_b2, o_rel);
}
